// FlashBackp_74783970558546
// MI455X (gfx1250) — matrix operand order repaired and checked by reading, not compiled, not run on hardware
//
#include <hip/hip_runtime.h>
#include <hip/hip_bf16.h>
#include <math.h>

#define BB 1024
#define SS 100
#define HH 256
#define LL 20000
#define UU 20000
#define ELN 500000
#define EUN 400000

typedef __attribute__((ext_vector_type(16))) __bf16 v16bf;
typedef __attribute__((ext_vector_type(8)))  float  v8f;

union Frag { v16bf v; unsigned short u[16]; uint4 q[2]; };

__device__ __forceinline__ unsigned short f2b(float f) {
  unsigned u = __float_as_uint(f);
  u += 0x7FFFu + ((u >> 16) & 1u);          // round-to-nearest-even to bf16
  return (unsigned short)(u >> 16);
}

// A fragment: 16x32 bf16 tile from row-major [m][k] matrix (ld in elements).
// Layout (ISA 7.12.2): lane m=lane&15; halves 0..7 -> K = 8*hi + i,
// halves 8..15 -> K = 16 + 8*hi + (i-8), hi = lane>>4.
template<typename PT>
__device__ __forceinline__ v16bf load_a(const PT* base, int ld, int m0, int k0) {
  const int lane = threadIdx.x & 31;
  const PT* p = base + (size_t)(m0 + (lane & 15)) * ld + k0 + ((lane >> 4) << 3);
  Frag f;
  f.q[0] = *(const uint4*)p;
  f.q[1] = *(const uint4*)(p + 16);
  return f.v;
}

// B fragment: 32x16 bf16 tile; element (k,n) stored row-major as M[n][k]
// (i.e. a W matrix used as x @ W^T). Lane n=lane&15; halves 0..7 -> K = 8*hi + i,
// halves 8..15 -> K = 16 + 8*hi + (i-8).
template<typename PT>
__device__ __forceinline__ v16bf load_b(const PT* base, int ld, int k0, int n0) {
  const int lane = threadIdx.x & 31;
  const PT* p = base + (size_t)(n0 + (lane & 15)) * ld + k0 + ((lane >> 4) << 3);
  Frag f;
  f.q[0] = *(const uint4*)p;
  f.q[1] = *(const uint4*)(p + 16);
  return f.v;
}

// ---------------- SpMM: one wave per COO edge, 8 floats/lane, atomic scatter.
__global__ void spmm_kernel(const int* __restrict__ rows, const int* __restrict__ cols,
                            const float* __restrict__ vals, const float* __restrict__ emb,
                            float* __restrict__ out, int nnz) {
  int wid  = (blockIdx.x * blockDim.x + threadIdx.x) >> 5;
  int lane = threadIdx.x & 31;
  if (wid >= nnz) return;
  int r = rows[wid], c = cols[wid];
  float v = vals[wid];
  const float4* src = (const float4*)(emb + (size_t)c * HH) + lane * 2;
  float* dst = out + (size_t)r * HH + lane * 8;
  float4 s0 = src[0], s1 = src[1];
  atomicAdd(dst + 0, v * s0.x); atomicAdd(dst + 1, v * s0.y);
  atomicAdd(dst + 2, v * s0.z); atomicAdd(dst + 3, v * s0.w);
  atomicAdd(dst + 4, v * s1.x); atomicAdd(dst + 5, v * s1.y);
  atomicAdd(dst + 6, v * s1.z); atomicAdd(dst + 7, v * s1.w);
}

// ---------------- similarity + decay weights + x_emb -> bf16 (time-major)
__global__ void simw_kernel(const int* __restrict__ fsm, const int* __restrict__ uid,
                            const float* __restrict__ td, const float* __restrict__ gd,
                            const int* __restrict__ length,
                            const float* __restrict__ enc_w, const float* __restrict__ ew_user,
                            unsigned short* __restrict__ x_bf,
                            float* __restrict__ w_j, float* __restrict__ denom) {
  int gw   = (blockIdx.x * blockDim.x + threadIdx.x) >> 5;
  int lane = threadIdx.x & 31;
  if (gw >= BB * SS) return;
  int b = gw / SS, t = gw % SS;
  int loc = fsm[b * SS + t];
  int u   = uid[b];
  const float* xr = enc_w  + (size_t)loc * HH + lane * 8;
  const float* ur = ew_user + (size_t)u  * HH + lane * 8;
  unsigned short* xo = x_bf + ((size_t)t * BB + b) * HH + lane * 8;
  float s = 0.f;
  #pragma unroll
  for (int i = 0; i < 8; ++i) {
    float x = xr[i];
    xo[i] = f2b(x);
    float d = ur[i] - x;
    s += d * d;
  }
  #pragma unroll
  for (int off = 16; off >= 1; off >>= 1) s += __shfl_xor(s, off, 32);
  if (lane == 0) {
    float sim = expf(-sqrtf(s));
    float tdv = td[b * SS + t], gdv = gd[b * SS + t];
    float aj = (cosf(tdv * 7.2722052166430399e-05f) + 1.f) * 0.5f
             * expf(-tdv * (0.1f / 86400.0f));
    float bj = expf(-gdv * 1000.0f);
    float w  = (aj * bj + 1e-10f) * sim;
    if (t >= length[b]) w = 0.f;
    w_j[b * SS + t] = w;
    atomicAdd(&denom[b], w);
  }
}

// ---------------- fp32 -> bf16 conversion
__global__ void cvt_kernel(const float* __restrict__ in, unsigned short* __restrict__ out, long n) {
  long i = (long)blockIdx.x * blockDim.x + threadIdx.x;
  if (i < n) out[i] = f2b(in[i]);
}

// ---------------- fused tanh-RNN + weighted-output accumulation.
// Block = 4 waves, 16 batch rows. Both weights resident in LDS (256 KB bf16),
// h double-buffered in LDS, 1 barrier per step. Each wave owns 64 output cols.
__global__ void __launch_bounds__(128) rnn_kernel(
    const unsigned short* __restrict__ x_bf,
    const float* __restrict__ Wih, const float* __restrict__ Whh,
    const float* __restrict__ bih, const float* __restrict__ bhh,
    const float* __restrict__ h0,  const float* __restrict__ w_j,
    const float* __restrict__ denom, float* __restrict__ out_w) {
  extern __shared__ unsigned short smem[];
  unsigned short* Wih_l = smem;                 // HH*HH bf16
  unsigned short* Whh_l = smem + HH * HH;       // HH*HH bf16
  unsigned short* hbuf  = smem + 2 * HH * HH;   // 2 * 16 * HH bf16

  const int row0  = blockIdx.x * 16;
  const int wave  = threadIdx.x >> 5;
  const int lane  = threadIdx.x & 31;
  const int hi    = lane >> 4;
  const int nbase = wave * 64;

  for (int i = threadIdx.x; i < HH * HH; i += blockDim.x) {
    Wih_l[i] = f2b(Wih[i]);
    Whh_l[i] = f2b(Whh[i]);
  }
  for (int i = threadIdx.x; i < 16 * HH; i += blockDim.x) {
    int m = i >> 8, k = i & (HH - 1);
    hbuf[i] = f2b(h0[(size_t)(row0 + m) * HH + k]);
  }
  __syncthreads();

  float bias[4];
  #pragma unroll
  for (int nt = 0; nt < 4; ++nt) {
    int n = nbase + nt * 16 + (lane & 15);
    bias[nt] = bih[n] + bhh[n];
  }
  v8f outacc[4];
  #pragma unroll
  for (int nt = 0; nt < 4; ++nt)
    #pragma unroll
    for (int r = 0; r < 8; ++r) outacc[nt][r] = 0.f;

  for (int t = 0; t < SS; ++t) {
    const unsigned short* xb = x_bf + ((size_t)t * BB + row0) * HH;
    const unsigned short* hc = hbuf + (t & 1) * (16 * HH);
    unsigned short*       hn = hbuf + ((t + 1) & 1) * (16 * HH);

    v8f acc[4];
    #pragma unroll
    for (int nt = 0; nt < 4; ++nt)
      #pragma unroll
      for (int r = 0; r < 8; ++r) acc[nt][r] = bias[nt];

    for (int kt = 0; kt < 8; ++kt) {
      v16bf ax = load_a(xb, HH, 0, kt * 32);
      v16bf ah = load_a(hc, HH, 0, kt * 32);
      #pragma unroll
      for (int nt = 0; nt < 4; ++nt) {
        v16bf bi = load_b(Wih_l, HH, kt * 32, nbase + nt * 16);
        acc[nt] = __builtin_amdgcn_wmma_f32_16x16x32_bf16(
            false, ax, false, bi, (short)0, acc[nt], false, false);
        v16bf bh = load_b(Whh_l, HH, kt * 32, nbase + nt * 16);
        acc[nt] = __builtin_amdgcn_wmma_f32_16x16x32_bf16(
            false, ah, false, bh, (short)0, acc[nt], false, false);
      }
    }
    #pragma unroll
    for (int nt = 0; nt < 4; ++nt) {
      int n = nbase + nt * 16 + (lane & 15);
      #pragma unroll
      for (int r = 0; r < 8; ++r) {
        int m = r + 8 * hi;
        float hv = tanhf(acc[nt][r]);
        hn[m * HH + n] = f2b(hv);
        outacc[nt][r] += w_j[(size_t)(row0 + m) * SS + t] * hv;
      }
    }
    __syncthreads();
  }
  #pragma unroll
  for (int nt = 0; nt < 4; ++nt) {
    int n = nbase + nt * 16 + (lane & 15);
    #pragma unroll
    for (int r = 0; r < 8; ++r) {
      int m = r + 8 * hi;
      out_w[(size_t)(row0 + m) * HH + n] = outacc[nt][r] / denom[row0 + m];
    }
  }
}

// ---------------- out_w2 = sum_t emb2[full_seq]*w / denom
__global__ void outw2_kernel(const int* __restrict__ fs, const float* __restrict__ emb2,
                             const float* __restrict__ w_j, const float* __restrict__ denom,
                             float* __restrict__ out_w2) {
  int i = blockIdx.x * blockDim.x + threadIdx.x;
  if (i >= BB * HH) return;
  int b = i >> 8, h = i & 255;
  float acc = 0.f;
  for (int t = 0; t < SS; ++t) {
    int loc = fs[b * SS + t];
    acc += emb2[(size_t)loc * HH + h] * w_j[b * SS + t];
  }
  out_w2[i] = acc / denom[b];
}

// ---------------- feat = concat(out_w, user_emb[uid], out_w2) as bf16
__global__ void feat_kernel(const float* __restrict__ out_w, const float* __restrict__ user_emb,
                            const int* __restrict__ uid, const float* __restrict__ out_w2,
                            unsigned short* __restrict__ feat) {
  int i = blockIdx.x * blockDim.x + threadIdx.x;
  if (i >= BB * 768) return;
  int b = i / 768, k = i % 768;
  float v;
  if (k < 256)       v = out_w[b * HH + k];
  else if (k < 512)  v = user_emb[(size_t)uid[b] * HH + (k - 256)];
  else               v = out_w2[b * HH + (k - 512)];
  feat[i] = f2b(v);
}

// ---------------- final GEMM: out[1024,20000] = feat[1024,768] @ W1^T + b1
// Wave tile: 16 x 80 (1 M-tile x 5 N-tiles). Block = 8 waves -> 128 x 80.
__global__ void __launch_bounds__(256) gemm_kernel(
    const unsigned short* __restrict__ feat, const unsigned short* __restrict__ w1b,
    const float* __restrict__ b1, float* __restrict__ out) {
  const int wave = threadIdx.x >> 5;
  const int lane = threadIdx.x & 31;
  const int hi   = lane >> 4;
  const int m0   = blockIdx.y * 128 + wave * 16;
  const int n0   = blockIdx.x * 80;

  v8f acc[5];
  #pragma unroll
  for (int nt = 0; nt < 5; ++nt) {
    float bv = b1[n0 + nt * 16 + (lane & 15)];
    #pragma unroll
    for (int r = 0; r < 8; ++r) acc[nt][r] = bv;
  }
  for (int kt = 0; kt < 24; ++kt) {
    v16bf a = load_a(feat, 768, m0, kt * 32);
    #pragma unroll
    for (int nt = 0; nt < 5; ++nt) {
      v16bf bfr = load_b(w1b, 768, kt * 32, n0 + nt * 16);
      acc[nt] = __builtin_amdgcn_wmma_f32_16x16x32_bf16(
          false, a, false, bfr, (short)0, acc[nt], false, false);
    }
  }
  #pragma unroll
  for (int nt = 0; nt < 5; ++nt) {
    int n = n0 + nt * 16 + (lane & 15);
    #pragma unroll
    for (int r = 0; r < 8; ++r) {
      int m = m0 + r + 8 * hi;
      out[(size_t)m * LL + n] = acc[nt][r];
    }
  }
}

extern "C" void kernel_launch(void* const* d_in, const int* in_sizes, int n_in,
                              void* d_out, int out_size, void* d_ws, size_t ws_size,
                              hipStream_t stream) {
  const int*   full_seq     = (const int*)  d_in[0];
  const int*   full_seq_map = (const int*)  d_in[1];
  const int*   length       = (const int*)  d_in[2];
  const int*   user_id      = (const int*)  d_in[3];
  const float* time_delta   = (const float*)d_in[4];
  const float* geo_delta    = (const float*)d_in[5];
  const float* enc_emb      = (const float*)d_in[6];
  const float* user_emb     = (const float*)d_in[7];
  const float* emb2         = (const float*)d_in[8];
  const int*   row_loc      = (const int*)  d_in[9];
  const int*   col_loc      = (const int*)  d_in[10];
  const float* vals_loc     = (const float*)d_in[11];
  const int*   row_usr      = (const int*)  d_in[12];
  const int*   col_usr      = (const int*)  d_in[13];
  const float* vals_usr     = (const float*)d_in[14];
  const float* W_ih         = (const float*)d_in[15];
  const float* W_hh         = (const float*)d_in[16];
  const float* b_ih         = (const float*)d_in[17];
  const float* b_hh         = (const float*)d_in[18];
  const float* W1           = (const float*)d_in[19];
  const float* b1           = (const float*)d_in[20];
  const float* h0           = (const float*)d_in[21];
  (void)in_sizes; (void)n_in; (void)out_size; (void)ws_size;

  char* ws = (char*)d_ws;                                   // 256B-aligned offsets
  float*          enc_w   = (float*)(ws + 0);               // L*H f32  = 20,480,000
  float*          ew_user = (float*)(ws + 20480000);        // U*H f32  = 20,480,000
  unsigned short* x_bf    = (unsigned short*)(ws + 40960000);  // S*B*H bf16 = 52,428,800
  float*          w_j     = (float*)(ws + 93388800);        // B*S f32
  float*          denom   = (float*)(ws + 93798400);        // B f32
  float*          out_w   = (float*)(ws + 93802496);        // B*H f32
  float*          out_w2  = (float*)(ws + 94851072);        // B*H f32
  unsigned short* feat    = (unsigned short*)(ws + 95899648);  // B*768 bf16
  unsigned short* w1b     = (unsigned short*)(ws + 97472512);  // L*768 bf16

  hipMemsetAsync(enc_w,   0, (size_t)LL * HH * sizeof(float), stream);
  hipMemsetAsync(ew_user, 0, (size_t)UU * HH * sizeof(float), stream);
  hipMemsetAsync(denom,   0, (size_t)BB * sizeof(float), stream);

  spmm_kernel<<<(ELN + 7) / 8, 256, 0, stream>>>(row_loc, col_loc, vals_loc, enc_emb, enc_w, ELN);
  spmm_kernel<<<(EUN + 7) / 8, 256, 0, stream>>>(row_usr, col_usr, vals_usr, enc_emb, ew_user, EUN);
  cvt_kernel<<<((long)LL * 768 + 255) / 256, 256, 0, stream>>>(W1, w1b, (long)LL * 768);
  simw_kernel<<<(BB * SS) / 8, 256, 0, stream>>>(full_seq_map, user_id, time_delta, geo_delta,
                                                 length, enc_w, ew_user, x_bf, w_j, denom);
  rnn_kernel<<<BB / 16, 128, (2 * HH * HH + 2 * 16 * HH) * sizeof(unsigned short), stream>>>(
      x_bf, W_ih, W_hh, b_ih, b_hh, h0, w_j, denom, out_w);
  outw2_kernel<<<(BB * HH) / 256, 256, 0, stream>>>(full_seq, emb2, w_j, denom, out_w2);
  feat_kernel<<<(BB * 768) / 256, 256, 0, stream>>>(out_w, user_emb, user_id, out_w2, feat);
  gemm_kernel<<<dim3(250, 8), 256, 0, stream>>>(feat, w1b, b1, (float*)d_out);
}
